// Dynamics_21053929685275
// MI455X (gfx1250) — matrix operand order repaired and checked by reading, not compiled, not run on hardware
//
#include <hip/hip_runtime.h>
#include <cstdint>
#include <cstddef>

typedef unsigned short u16;
typedef __attribute__((ext_vector_type(16))) __bf16 v16bf;
typedef __attribute__((ext_vector_type(8)))  float  v8f;
typedef int vsi4 __attribute__((__vector_size__(4 * sizeof(int))));   // builtin's int4 payload type

#define BB    16384
#define DD    256
#define HH    2048
#define OUTN  255
#define ALPHA_C  0.01f
#define REHU_D_C 0.01f
#define EPS_C    1e-5f
#define INV_H    (1.0f / 2048.0f)

// CDNA5 async global->LDS copies (ASYNCcnt path), if this toolchain declares them.
#if defined(__has_builtin)
#if __has_builtin(__builtin_amdgcn_global_load_async_to_lds_b128)
#define HAVE_ASYNC_LDS 1
#endif
#endif
#ifndef HAVE_ASYNC_LDS
#define HAVE_ASYNC_LDS 0
#endif

#if HAVE_ASYNC_LDS
#if __has_builtin(__builtin_amdgcn_s_wait_asynccnt)
#define WAIT_ASYNC(n) __builtin_amdgcn_s_wait_asynccnt(n)
#else
#define WAIT_ASYNC(n) asm volatile("s_wait_asynccnt %0" :: "i"(n) : "memory")
#endif
#else
#define WAIT_ASYNC(n) ((void)0)
#endif

#define AS1 __attribute__((address_space(1)))
#define AS3 __attribute__((address_space(3)))

// ---------- scalar helpers ----------
__device__ __forceinline__ u16 f2bf(float f) {
    uint32_t u = __float_as_uint(f);
    uint32_t r = u + 0x7FFFu + ((u >> 16) & 1u);   // round-to-nearest-even
    return (u16)(r >> 16);
}
__device__ __forceinline__ float bf2f(u16 h) {
    return __uint_as_float(((uint32_t)h) << 16);
}
__device__ __forceinline__ float rehu_f(float x, float d) {
    float q = x * fabsf(x) / (2.0f * d);           // sign(x)*x^2/(2d)
    float c = fminf(fmaxf(q, 0.0f), 0.5f * d);
    return fmaxf(c, x - 0.5f * d);
}
__device__ __forceinline__ float rehu_grad(float x, float d) {
    return fminf(fmaxf(x / d, 0.0f), 1.0f);
}
__device__ __forceinline__ float softplus_f(float x) {
    return (x > 15.0f) ? x : log1pf(expf(x));
}

__device__ __forceinline__ float blockReduceSum(float v, float* s) {
    int t = threadIdx.x;
    s[t] = v; __syncthreads();
    #pragma unroll
    for (int off = 128; off > 0; off >>= 1) {
        if (t < off) s[t] += s[t + off];
        __syncthreads();
    }
    float r = s[0]; __syncthreads();
    return r;
}

// ---------- conversion / prep kernels ----------
__global__ __launch_bounds__(256) void k_cvt_bf16(const float* __restrict__ src, u16* __restrict__ dst, int n) {
    int i = blockIdx.x * 256 + threadIdx.x;
    if (i < n) dst[i] = f2bf(src[i]);
}
__global__ __launch_bounds__(256) void k_softplus_bf16(const float* __restrict__ src, u16* __restrict__ dst, int n) {
    int i = blockIdx.x * 256 + threadIdx.x;
    if (i < n) dst[i] = f2bf(softplus_f(src[i]));
}
__global__ __launch_bounds__(256) void k_softplus_f32(const float* __restrict__ src, float* __restrict__ dst, int n) {
    int i = blockIdx.x * 256 + threadIdx.x;
    if (i < n) dst[i] = softplus_f(src[i]);
}
// fW2 padded to [256,H]: row 255 = 0; bias padded so fx[:,255] = 1 automatically
__global__ __launch_bounds__(256) void k_build_fW2(const float* __restrict__ fW2, u16* __restrict__ dst) {
    int i = blockIdx.x * 256 + threadIdx.x;   // over 256*H
    if (i >= 256 * HH) return;
    int r = i >> 11, k = i & (HH - 1);
    dst[i] = (r < OUTN) ? f2bf(fW2[r * HH + k]) : (u16)0;
}
__global__ __launch_bounds__(256) void k_build_fb2(const float* __restrict__ fb2, float* __restrict__ dst) {
    int i = threadIdx.x;
    dst[i] = (i < OUTN) ? fb2[i] : 1.0f;
}

// ---------- WMMA bf16 GEMM, double-buffered async tiles, fused epilogues ----------
// MODE_T==1 : W stored [N,K] row-major (compute A@W^T)
// MODE_T==0 : W stored [K,N] row-major (compute A@W)
// EPI: 0=f32 C store; 1=bf16 relu(v+bias); 2=z2+g2; 3=f32 v+bias (fx); 4=g1=v*aux; 5=z1+r1
union Frag32 { uint4 q[2]; v16bf v; };

template<int MODE_T, int ACCUM, int EPI>
__global__ __launch_bounds__(256) void gemm_bf16(const u16* __restrict__ A, const u16* __restrict__ W,
                                                 float* __restrict__ C, int M, int N, int K, float alpha,
                                                 const float* __restrict__ bias, const float* __restrict__ bias2,
                                                 u16* __restrict__ out0, u16* __restrict__ out1,
                                                 const u16* __restrict__ aux) {
    constexpr int LSTR = 40;                       // 32 + 8 bf16 padding, keeps 16B alignment
    constexpr int TILE = 128 * LSTR;
    __shared__ __align__(16) u16 sA[2 * TILE];
    __shared__ __align__(16) u16 sB[2 * TILE];

    const int tid  = threadIdx.x;
    const int lane = tid & 31;
    const int wave = tid >> 5;
    const int wm = wave >> 1;                      // 0..3 -> 32-row strip
    const int wn = wave & 1;                       // 0..1 -> 64-col strip
    const int bM = blockIdx.y * 128;
    const int bN = blockIdx.x * 128;

    v8f acc[2][4];
    v8f vzero = {};
    #pragma unroll
    for (int i = 0; i < 2; ++i)
        #pragma unroll
        for (int j = 0; j < 4; ++j) acc[i][j] = vzero;

    const int lc    = lane & 15;
    const int aksel = (lane < 16) ? 0 : 8;         // A k-chunk select (ushort offset)
    const int bksel = (lane < 16) ? 0 : 8;         // B k-chunk select

    // issue one 128x32 A tile + 32-wide B tile at k0 into buffer bsel
    auto issue_tile = [&](int k0, int bsel) {
        u16* dA = sA + bsel * TILE;
        u16* dB = sB + bsel * TILE;
        #pragma unroll
        for (int i = 0; i < 2; ++i) {
            int c = tid + i * 256;                 // 0..511
            int r = c >> 2, cc = c & 3;
            const u16* src = A + (size_t)(bM + r) * K + k0 + cc * 8;
            __builtin_prefetch(src + 64, 0, 1);    // speculative, 2 tiles ahead
#if HAVE_ASYNC_LDS
            __builtin_amdgcn_global_load_async_to_lds_b128(
                (AS1 vsi4*)const_cast<u16*>(src), (AS3 vsi4*)(dA + r * LSTR + cc * 8), 0, 0);
#else
            *(uint4*)(dA + r * LSTR + cc * 8) = *(const uint4*)src;
#endif
        }
        if (MODE_T) {
            #pragma unroll
            for (int i = 0; i < 2; ++i) {
                int c = tid + i * 256;
                int r = c >> 2, cc = c & 3;
                const u16* src = W + (size_t)(bN + r) * K + k0 + cc * 8;
                __builtin_prefetch(src + 64, 0, 1);
#if HAVE_ASYNC_LDS
                __builtin_amdgcn_global_load_async_to_lds_b128(
                    (AS1 vsi4*)const_cast<u16*>(src), (AS3 vsi4*)(dB + r * LSTR + cc * 8), 0, 0);
#else
                *(uint4*)(dB + r * LSTR + cc * 8) = *(const uint4*)src;
#endif
            }
        } else {
            #pragma unroll
            for (int i = 0; i < 2; ++i) {
                int c = tid + i * 256;
                int kk = c >> 4;                   // 0..31
                int nn = (c & 15) * 8;             // 0..120
                const u16* src = W + (size_t)(k0 + kk) * N + bN + nn;
                __builtin_prefetch(src + (size_t)64 * N, 0, 1);
                #pragma unroll
                for (int e = 0; e < 8; ++e) dB[(nn + e) * LSTR + kk] = src[e];
            }
        }
    };

    auto compute_tile = [&](int cur) {
        const u16* cA = sA + cur * TILE;
        const u16* cB = sB + cur * TILE;
        Frag32 af[2], bfr[4];
        #pragma unroll
        for (int i = 0; i < 2; ++i) {
            const u16* base = cA + (wm * 32 + i * 16 + lc) * LSTR;
            af[i].q[0] = *(const uint4*)(base + aksel);
            af[i].q[1] = *(const uint4*)(base + aksel + 16);
        }
        #pragma unroll
        for (int j = 0; j < 4; ++j) {
            const u16* base = cB + (wn * 64 + j * 16 + lc) * LSTR + bksel;
            bfr[j].q[0] = *(const uint4*)(base);
            bfr[j].q[1] = *(const uint4*)(base + 16);
        }
        #pragma unroll
        for (int i = 0; i < 2; ++i)
            #pragma unroll
            for (int j = 0; j < 4; ++j)
                acc[i][j] = __builtin_amdgcn_wmma_f32_16x16x32_bf16(
                    false, af[i].v, false, bfr[j].v, (short)0, acc[i][j], false, false);
    };

    const int NT = K >> 5;
    issue_tile(0, 0);

    // steady state: branch-free body, next-tile DMA overlaps current-tile WMMA
    for (int kt = 0; kt < NT - 1; ++kt) {
        const int cur = kt & 1;
        __syncthreads();                                   // back buffer free (prev compute done)
        issue_tile((kt + 1) << 5, cur ^ 1);
        WAIT_ASYNC(MODE_T ? 4 : 2);                        // tile kt landed; tile kt+1 still in flight
        __syncthreads();                                   // tile kt visible to all waves
        compute_tile(cur);
    }
    // peeled last tile
    __syncthreads();
    WAIT_ASYNC(0);
    __syncthreads();
    compute_tile((NT - 1) & 1);

    // ---- fused epilogue ----
    const int rowb = bM + wm * 32;
    const int colb = bN + wn * 64;
    const int lhalf = (lane < 16) ? 0 : 8;
    #pragma unroll
    for (int i = 0; i < 2; ++i)
        #pragma unroll
        for (int j = 0; j < 4; ++j)
            #pragma unroll
            for (int r = 0; r < 8; ++r) {
                int row = rowb + i * 16 + r + lhalf;
                int col = colb + j * 16 + lc;
                size_t off = (size_t)row * N + col;
                float v = acc[i][j][r] * alpha;
                if (ACCUM) v += __builtin_nontemporal_load(C + off);   // staging read-once
                if (EPI == 0) {
                    __builtin_nontemporal_store(v, C + off);           // staging write-once
                } else if (EPI == 1) {
                    out0[off] = f2bf(fmaxf(v + bias[col], 0.0f));
                } else if (EPI == 2) {
                    float a = v + bias[col];
                    out0[off] = f2bf(rehu_f(a, REHU_D_C));
                    out1[off] = f2bf(bias2[col] * rehu_grad(a, REHU_D_C) * INV_H);
                } else if (EPI == 3) {
                    C[off] = v + bias[col];
                } else if (EPI == 4) {
                    out0[off] = f2bf(v * bf2f(aux[off]));
                } else if (EPI == 5) {
                    float a = v + bias[col];
                    out0[off] = f2bf(rehu_f(a, REHU_D_C));
                    out1[off] = f2bf(rehu_grad(a, REHU_D_C));
                }
            }
}

// ---------- icnn(0) (tiny, f32 path for accuracy) ----------
__global__ __launch_bounds__(256) void k_z1_0(const float* __restrict__ vb0, float* __restrict__ z1_0) {
    int i = blockIdx.x * 256 + threadIdx.x;
    if (i < HH) z1_0[i] = rehu_f(vb0[i], REHU_D_C);
}
__global__ __launch_bounds__(256) void k_a2_0(const float* __restrict__ vU0, const float* __restrict__ vb1,
                                              const float* __restrict__ z1_0, float* __restrict__ z2_0) {
    __shared__ float sred[256];
    int j = blockIdx.x;
    float s = 0.0f;
    for (int i = threadIdx.x; i < HH; i += 256)
        s += softplus_f(vU0[(size_t)j * HH + i]) * z1_0[i];
    s = blockReduceSum(s, sred);
    if (threadIdx.x == 0) z2_0[j] = rehu_f(vb1[j] + s * INV_H, REHU_D_C);
}
__global__ __launch_bounds__(256) void k_icnn0(const float* __restrict__ z2_0, const float* __restrict__ spU1,
                                               const float* __restrict__ vb2, float* __restrict__ icnn0) {
    __shared__ float sred[256];
    float s = 0.0f;
    for (int i = threadIdx.x; i < HH; i += 256) s += z2_0[i] * spU1[i];
    s = blockReduceSum(s, sred);
    if (threadIdx.x == 0) icnn0[0] = vb2[0] + s * INV_H;
}

// ---------- per-row: s = icnn(x)-icnn0, rp = rehu'(s,1), Vx ----------
__global__ __launch_bounds__(256) void k_rowV(const float* __restrict__ x, const float* __restrict__ vW2,
                                              const float* __restrict__ vb2, const float* __restrict__ spU1,
                                              const u16* __restrict__ z2b, const float* __restrict__ icnn0,
                                              float* __restrict__ rp, float* __restrict__ Vx) {
    __shared__ float sred[256];
    int b = blockIdx.x, d = threadIdx.x;
    float xv = x[(size_t)b * DD + d];
    float dx = blockReduceSum(xv * vW2[d], sred);
    float ss = blockReduceSum(xv * xv, sred);
    float dzp = 0.0f;
    for (int j = d; j < HH; j += 256) dzp += bf2f(z2b[(size_t)b * HH + j]) * spU1[j];
    float dz = blockReduceSum(dzp, sred);
    if (d == 0) {
        float s = dx + vb2[0] + dz * INV_H - icnn0[0];
        rp[b] = fminf(fmaxf(s, 0.0f), 1.0f);            // rehu'(s, d=1)
        Vx[b] = rehu_f(s, 1.0f) + EPS_C * ss;
    }
}

// ---------- final projection per row ----------
__global__ __launch_bounds__(256) void k_final(const float* __restrict__ x, const float* __restrict__ vW2,
                                               const float* __restrict__ fx, const float* __restrict__ gvpart,
                                               const float* __restrict__ rp, const float* __restrict__ Vx,
                                               float* __restrict__ out) {
    __shared__ float sred[256];
    int b = blockIdx.x, d = threadIdx.x;
    size_t o = (size_t)b * DD + d;
    float gv  = rp[b] * (vW2[d] + gvpart[o]) + 2.0f * EPS_C * x[o];
    float fxv = fx[o];
    float s1 = blockReduceSum(gv * fxv, sred);
    float s2 = blockReduceSum(gv * gv, sred);
    float num  = fmaxf(s1 + ALPHA_C * Vx[b], 0.0f);
    float coef = num / s2;
    float rv = fxv - gv * coef;
    if (d < OUTN) out[(size_t)b * OUTN + d] = rv;
}

// ---------- host orchestration ----------
extern "C" void kernel_launch(void* const* d_in, const int* in_sizes, int n_in,
                              void* d_out, int out_size, void* d_ws, size_t ws_size,
                              hipStream_t stream) {
    const float* x   = (const float*)d_in[0];
    const float* fW0 = (const float*)d_in[1];
    const float* fb0 = (const float*)d_in[2];
    const float* fW1 = (const float*)d_in[3];
    const float* fb1 = (const float*)d_in[4];
    const float* fW2 = (const float*)d_in[5];
    const float* fb2 = (const float*)d_in[6];
    const float* vW0 = (const float*)d_in[7];
    const float* vb0 = (const float*)d_in[8];
    const float* vW1 = (const float*)d_in[9];
    const float* vb1 = (const float*)d_in[10];
    const float* vW2 = (const float*)d_in[11];
    const float* vb2 = (const float*)d_in[12];
    const float* vU0 = (const float*)d_in[13];
    const float* vU1 = (const float*)d_in[14];
    float* out = (float*)d_out;

    char* p = (char*)d_ws;
    auto alloc = [&](size_t bytes) -> void* {
        void* r = (void*)p;
        p += (bytes + 255) & ~(size_t)255;
        return r;
    };

    u16* fW0b  = (u16*)alloc((size_t)HH * DD * 2);
    u16* fW1b  = (u16*)alloc((size_t)HH * HH * 2);
    u16* fW2b  = (u16*)alloc((size_t)256 * HH * 2);
    u16* vW0b  = (u16*)alloc((size_t)HH * DD * 2);
    u16* vW1b  = (u16*)alloc((size_t)HH * DD * 2);
    u16* spU0b = (u16*)alloc((size_t)HH * HH * 2);
    u16* xb    = (u16*)alloc((size_t)BB * DD * 2);
    float* spU1f  = (float*)alloc(HH * 4);
    float* fb2pad = (float*)alloc(256 * 4);
    float* z1_0   = (float*)alloc(HH * 4);
    float* z2_0   = (float*)alloc(HH * 4);
    float* icnn0  = (float*)alloc(256);
    float* tf32   = (float*)alloc((size_t)BB * HH * 4);    // f32 staging (x@vW1^T only)
    float* fxbuf  = (float*)alloc((size_t)BB * 256 * 4);
    float* gvpart = (float*)alloc((size_t)BB * 256 * 4);
    float* rp     = (float*)alloc((size_t)BB * 4);
    float* Vxb    = (float*)alloc((size_t)BB * 4);
    u16* A0 = (u16*)alloc((size_t)BB * HH * 2);            // h1 -> g2
    u16* A1 = (u16*)alloc((size_t)BB * HH * 2);            // h2 -> g1
    u16* A2 = (u16*)alloc((size_t)BB * HH * 2);            // z1
    u16* A3 = (u16*)alloc((size_t)BB * HH * 2);            // r1
    u16* A4 = (u16*)alloc((size_t)BB * HH * 2);            // z2
    (void)ws_size; (void)n_in; (void)in_sizes; (void)out_size;

    dim3 blk(256);
    auto g1d = [](int n) { return dim3((unsigned)((n + 255) / 256)); };

    // weight / input prep
    k_cvt_bf16<<<g1d(HH * DD), blk, 0, stream>>>(fW0, fW0b, HH * DD);
    k_cvt_bf16<<<g1d(HH * HH), blk, 0, stream>>>(fW1, fW1b, HH * HH);
    k_build_fW2<<<g1d(256 * HH), blk, 0, stream>>>(fW2, fW2b);
    k_build_fb2<<<dim3(1), blk, 0, stream>>>(fb2, fb2pad);
    k_cvt_bf16<<<g1d(HH * DD), blk, 0, stream>>>(vW0, vW0b, HH * DD);
    k_cvt_bf16<<<g1d(HH * DD), blk, 0, stream>>>(vW1, vW1b, HH * DD);
    k_softplus_bf16<<<g1d(HH * HH), blk, 0, stream>>>(vU0, spU0b, HH * HH);
    k_softplus_f32<<<g1d(HH), blk, 0, stream>>>(vU1, spU1f, HH);
    k_cvt_bf16<<<g1d(BB * DD), blk, 0, stream>>>(x, xb, BB * DD);

    // icnn(0) constant
    k_z1_0<<<g1d(HH), blk, 0, stream>>>(vb0, z1_0);
    k_a2_0<<<dim3(HH), blk, 0, stream>>>(vU0, vb1, z1_0, z2_0);
    k_icnn0<<<dim3(1), blk, 0, stream>>>(z2_0, spU1f, vb2, icnn0);

    dim3 gH(HH / 128, BB / 128);   // N=2048 tiles
    dim3 gD(256 / 128, BB / 128);  // N=256 tiles

    // fhat MLP (fused bias+relu epilogues, bf16 out)
    gemm_bf16<1, 0, 1><<<gH, blk, 0, stream>>>(xb, fW0b, tf32, BB, HH, DD, 1.0f,
                                               fb0, nullptr, A0, nullptr, nullptr);       // h1
    gemm_bf16<1, 0, 1><<<gH, blk, 0, stream>>>(A0, fW1b, tf32, BB, HH, HH, 1.0f,
                                               fb1, nullptr, A1, nullptr, nullptr);       // h2
    gemm_bf16<1, 0, 3><<<gD, blk, 0, stream>>>(A1, fW2b, fxbuf, BB, 256, HH, 1.0f,
                                               fb2pad, nullptr, nullptr, nullptr, nullptr); // fx (col 255 == 1)

    // ICNN forward
    gemm_bf16<1, 0, 5><<<gH, blk, 0, stream>>>(xb, vW0b, tf32, BB, HH, DD, 1.0f,
                                               vb0, nullptr, A2, A3, nullptr);            // z1, r1
    gemm_bf16<1, 0, 0><<<gH, blk, 0, stream>>>(xb, vW1b, tf32, BB, HH, DD, 1.0f,
                                               nullptr, nullptr, nullptr, nullptr, nullptr); // x@vW1^T (f32)
    gemm_bf16<1, 1, 2><<<gH, blk, 0, stream>>>(A2, spU0b, tf32, BB, HH, HH, INV_H,
                                               vb1, spU1f, A4, A0, nullptr);              // z2, g2

    // per-row V, rp
    k_rowV<<<dim3(BB), blk, 0, stream>>>(x, vW2, vb2, spU1f, A4, icnn0, rp, Vxb);

    // ICNN backward (analytic gradient)
    gemm_bf16<0, 0, 4><<<gH, blk, 0, stream>>>(A0, spU0b, tf32, BB, HH, HH, INV_H,
                                               nullptr, nullptr, A1, nullptr, A3);        // g1 = (G2@spU0)/H * r1
    gemm_bf16<0, 0, 0><<<gD, blk, 0, stream>>>(A0, vW1b, gvpart, BB, DD, HH, 1.0f,
                                               nullptr, nullptr, nullptr, nullptr, nullptr); // G2@vW1
    gemm_bf16<0, 1, 0><<<gD, blk, 0, stream>>>(A1, vW0b, gvpart, BB, DD, HH, 1.0f,
                                               nullptr, nullptr, nullptr, nullptr, nullptr); // += G1@vW0

    // final projection
    k_final<<<dim3(BB), blk, 0, stream>>>(x, vW2, fxbuf, gvpart, rp, Vxb, out);
}
